// Consolidation_24283745092289
// MI455X (gfx1250) — matrix operand order repaired and checked by reading, not compiled, not run on hardware
//
#include <hip/hip_runtime.h>

// ---------------- types / helpers ----------------
typedef __bf16 v16bf __attribute__((ext_vector_type(16)));
typedef float  v8f   __attribute__((ext_vector_type(8)));

struct Pack32 { uint4 a; uint4 b; };   // 32 bytes == v16bf

__device__ __forceinline__ unsigned short f2bf(float f) {
    unsigned int u = __builtin_bit_cast(unsigned int, f);
    u += 0x7FFFu + ((u >> 16) & 1u);           // round-to-nearest-even
    return (unsigned short)(u >> 16);
}
__device__ __forceinline__ float bf2f(unsigned short h) {
    return __builtin_bit_cast(float, (unsigned int)h << 16);
}

// A-matrix fragment (16x32 bf16, M x K). Per ISA 7.12.2:
// lanes 0-15 : row M=l, K = k0+0..7 (v0-3) and k0+16..23 (v4-7)
// lanes 16-31: row M=l, K = k0+8..15 and k0+24..31
__device__ __forceinline__ v16bf load_a_frag(const unsigned short* A, int lda,
                                             int row0, int k0, int lane) {
    const int l = lane & 15, hi = lane >> 4;
    const unsigned short* p = A + (size_t)(row0 + l) * lda + k0 + hi * 8;
    Pack32 t;
    t.a = *(const uint4*)(p);
    t.b = *(const uint4*)(p + 16);
    return __builtin_bit_cast(v16bf, t);
}
// B-matrix fragment (32x16 bf16, K x N) where B[k][n] = W[n][k] (W row-major [N,K]):
// lanes 0-15 : col N=l, K = k0+0..7 and k0+16..23 ; lanes 16-31: col N=l, K = k0+8..15 and k0+24..31
__device__ __forceinline__ v16bf load_b_frag(const unsigned short* W, int ldb,
                                             int col0, int k0, int lane) {
    const int l = lane & 15, hi = lane >> 4;
    const unsigned short* p = W + (size_t)(col0 + l) * ldb + k0 + hi * 8;
    Pack32 t;
    t.a = *(const uint4*)(p);
    t.b = *(const uint4*)(p + 16);
    return __builtin_bit_cast(v16bf, t);
}

// 64x32 C tile per wave: 4 M-frags x 2 N-frags, 8 f32 accumulators.
__device__ __forceinline__ void wmma_tile_64x32(const unsigned short* A, const unsigned short* W,
                                                int lda, int ldb, int tm, int tn, int K,
                                                int lane, v8f acc[4][2]) {
    for (int k0 = 0; k0 < K; k0 += 32) {
        v16bf a0 = load_a_frag(A, lda, tm +  0, k0, lane);
        v16bf a1 = load_a_frag(A, lda, tm + 16, k0, lane);
        v16bf a2 = load_a_frag(A, lda, tm + 32, k0, lane);
        v16bf a3 = load_a_frag(A, lda, tm + 48, k0, lane);
        v16bf b0 = load_b_frag(W, ldb, tn +  0, k0, lane);
        v16bf b1 = load_b_frag(W, ldb, tn + 16, k0, lane);
        acc[0][0] = __builtin_amdgcn_wmma_f32_16x16x32_bf16(false, a0, false, b0, (short)0, acc[0][0], false, false);
        acc[0][1] = __builtin_amdgcn_wmma_f32_16x16x32_bf16(false, a0, false, b1, (short)0, acc[0][1], false, false);
        acc[1][0] = __builtin_amdgcn_wmma_f32_16x16x32_bf16(false, a1, false, b0, (short)0, acc[1][0], false, false);
        acc[1][1] = __builtin_amdgcn_wmma_f32_16x16x32_bf16(false, a1, false, b1, (short)0, acc[1][1], false, false);
        acc[2][0] = __builtin_amdgcn_wmma_f32_16x16x32_bf16(false, a2, false, b0, (short)0, acc[2][0], false, false);
        acc[2][1] = __builtin_amdgcn_wmma_f32_16x16x32_bf16(false, a2, false, b1, (short)0, acc[2][1], false, false);
        acc[3][0] = __builtin_amdgcn_wmma_f32_16x16x32_bf16(false, a3, false, b0, (short)0, acc[3][0], false, false);
        acc[3][1] = __builtin_amdgcn_wmma_f32_16x16x32_bf16(false, a3, false, b1, (short)0, acc[3][1], false, false);
    }
}

// ---------------- kernels ----------------
__global__ void f32_to_bf16_kernel(const float* __restrict__ src,
                                   unsigned short* __restrict__ dst, int n) {
    int i = blockIdx.x * blockDim.x + threadIdx.x;
    if (i < n) dst[i] = f2bf(src[i]);
}

// C[m][n] = sum_k A[m][k] * W[n][k], C f32, A/W bf16 row-major with K contiguous.
__global__ void gemm_bf16_f32(const unsigned short* __restrict__ A,
                              const unsigned short* __restrict__ W,
                              float* __restrict__ C, int M, int N, int K) {
    const int lane = threadIdx.x & 31;
    const int wave = (blockIdx.x * blockDim.x + threadIdx.x) >> 5;
    const int tilesN = N >> 5;
    const int tm = (wave / tilesN) << 6;
    const int tn = (wave % tilesN) << 5;
    if (tm >= M) return;                       // wave-uniform
    v8f acc[4][2] = {};
    wmma_tile_64x32(A, W, K, K, tm, tn, K, lane, acc);
    const int l = lane & 15, hi = lane >> 4;
    #pragma unroll
    for (int i = 0; i < 4; ++i)
        #pragma unroll
        for (int j = 0; j < 2; ++j)
            #pragma unroll
            for (int r = 0; r < 8; ++r)
                C[(size_t)(tm + i * 16 + hi * 8 + r) * N + (tn + j * 16 + l)] = acc[i][j][r];
}

// Batched A = (q @ g^T) * scale, bf16 out. batch = t*8+b over 64; M=N=1024, K=512.
__global__ void gemm_qg_bf16(const unsigned short* __restrict__ Q,
                             const unsigned short* __restrict__ G,
                             unsigned short* __restrict__ Aout, float scale) {
    const int lane = threadIdx.x & 31;
    const int wave = (blockIdx.x * blockDim.x + threadIdx.x) >> 5;
    const int WPB = (1024 / 64) * (1024 / 32);  // 512 waves per batch
    const int batch = wave / WPB;
    const int r = wave % WPB;
    const int b = batch & 7;
    const int tm = (r >> 5) << 6;
    const int tn = (r & 31) << 5;
    const unsigned short* A  = Q + (size_t)batch * 1024 * 512;
    const unsigned short* W  = G + (size_t)b * 1024 * 512;
    unsigned short* Cb = Aout + (size_t)batch * 1024 * 1024;
    v8f acc[4][2] = {};
    wmma_tile_64x32(A, W, 512, 512, tm, tn, 512, lane, acc);
    const int l = lane & 15, hi = lane >> 4;
    #pragma unroll
    for (int i = 0; i < 4; ++i)
        #pragma unroll
        for (int j = 0; j < 2; ++j)
            #pragma unroll
            for (int r2 = 0; r2 < 8; ++r2)
                Cb[(size_t)(tm + i * 16 + hi * 8 + r2) * 1024 + (tn + j * 16 + l)] =
                    f2bf(acc[i][j][r2] * scale);
}

// BN(eval) + LIF over T, mean over T -> g (f32 + bf16). j over B*Nkv*D.
__global__ void gate_lif_kernel(const float* __restrict__ Y,
                                const float* __restrict__ gamma, const float* __restrict__ beta,
                                const float* __restrict__ rmean, const float* __restrict__ rvar,
                                float* __restrict__ g, unsigned short* __restrict__ gb) {
    const int j = blockIdx.x * blockDim.x + threadIdx.x;   // 4,194,304
    const size_t S = 8u * 1024u * 512u;                    // stride per t
    const int e = j & 511;
    const float sc = gamma[e] * rsqrtf(rvar[e] + 1e-5f);
    const float sh = beta[e] - rmean[e] * sc;
    float v = 0.f, acc = 0.f;
    #pragma unroll
    for (int t = 0; t < 8; ++t) {
        float y = Y[(size_t)t * S + j] * sc + sh;
        float h = 0.5f * (v + y);                 // v + (x-v)/tau, tau=2
        float s = (h >= 1.f) ? 1.f : 0.f;         // H(h-1)
        v = h * (1.f - s);                        // hard reset
        acc += s;
    }
    float gv = acc * 0.125f;
    g[j] = gv;
    gb[j] = f2bf(gv);
}

// BN(eval) + LIF over T -> spikes (f32 out). j over B*Nq*D.
__global__ void proj_lif_kernel(const float* __restrict__ Y,
                                const float* __restrict__ gamma, const float* __restrict__ beta,
                                const float* __restrict__ rmean, const float* __restrict__ rvar,
                                float* __restrict__ out) {
    const int j = blockIdx.x * blockDim.x + threadIdx.x;
    const size_t S = 8u * 1024u * 512u;
    const int e = j & 511;
    const float sc = gamma[e] * rsqrtf(rvar[e] + 1e-5f);
    const float sh = beta[e] - rmean[e] * sc;
    float v = 0.f;
    #pragma unroll
    for (int t = 0; t < 8; ++t) {
        float y = Y[(size_t)t * S + j] * sc + sh;
        float h = 0.5f * (v + y);
        float s = (h >= 1.f) ? 1.f : 0.f;
        v = h * (1.f - s);
        out[(size_t)t * S + j] = s;
    }
}

// One wave per A row: top-4 (jax tie-break: smaller index on equal value),
// then update[row,:] = sum_j a_j * g[b, idx_j, :], written bf16.
__global__ void topk_update_kernel(const unsigned short* __restrict__ Amat,
                                   const float* __restrict__ g,
                                   unsigned short* __restrict__ upd) {
    const int lane = threadIdx.x & 31;
    const int wv = threadIdx.x >> 5;                 // 8 waves / block
    const int row = blockIdx.x * 8 + wv;             // 65536 rows = (t*8+b)*1024 + q
    const int b = (row >> 10) & 7;
    const unsigned short* Ar = Amat + (size_t)row * 1024;

    float tv0 = -3.402823466e38f, tv1 = tv0, tv2 = tv0, tv3 = tv0;
    int   ti0 = 0x7FFFFFFF, ti1 = ti0, ti2 = ti0, ti3 = ti0;
    for (int n = lane; n < 1024; n += 32) {
        float val = bf2f(Ar[n]);
        if (val > tv0)      { tv3=tv2;ti3=ti2; tv2=tv1;ti2=ti1; tv1=tv0;ti1=ti0; tv0=val;ti0=n; }
        else if (val > tv1) { tv3=tv2;ti3=ti2; tv2=tv1;ti2=ti1; tv1=val;ti1=n; }
        else if (val > tv2) { tv3=tv2;ti3=ti2; tv2=val;ti2=n; }
        else if (val > tv3) { tv3=val;ti3=n; }
    }

    __shared__ float sval[8][128];
    __shared__ int   sidx[8][128];
    __shared__ float pv[8][4];
    __shared__ int   pi[8][4];
    sval[wv][lane * 4 + 0] = tv0; sidx[wv][lane * 4 + 0] = ti0;
    sval[wv][lane * 4 + 1] = tv1; sidx[wv][lane * 4 + 1] = ti1;
    sval[wv][lane * 4 + 2] = tv2; sidx[wv][lane * 4 + 2] = ti2;
    sval[wv][lane * 4 + 3] = tv3; sidx[wv][lane * 4 + 3] = ti3;
    __syncthreads();

    if (lane == 0) {
        for (int k = 0; k < 4; ++k) {
            float bv = -3.402823466e38f; int bi = 0x7FFFFFFF; int bpos = 0;
            for (int c = 0; c < 128; ++c) {
                float vv = sval[wv][c]; int ix = sidx[wv][c];
                if (vv > bv || (vv == bv && ix < bi)) { bv = vv; bi = ix; bpos = c; }
            }
            pv[wv][k] = bv; pi[wv][k] = bi;
            sval[wv][bpos] = -3.402823466e38f;
        }
    }
    __syncthreads();

    const float a0 = pv[wv][0], a1 = pv[wv][1], a2 = pv[wv][2], a3 = pv[wv][3];
    const float* g0 = g + ((size_t)(b << 10) + pi[wv][0]) * 512;
    const float* g1 = g + ((size_t)(b << 10) + pi[wv][1]) * 512;
    const float* g2 = g + ((size_t)(b << 10) + pi[wv][2]) * 512;
    const float* g3 = g + ((size_t)(b << 10) + pi[wv][3]) * 512;
    unsigned short* ur = upd + (size_t)row * 512;
    const int d0 = lane * 16;
    #pragma unroll
    for (int d = 0; d < 16; ++d) {
        int dd = d0 + d;
        float s = a0 * g0[dd] + a1 * g1[dd] + a2 * g2[dd] + a3 * g3[dd];
        ur[dd] = f2bf(s);
    }
}

// ---------------- workspace layout (bytes) ----------------
// [0          , 64Mi)  q_bf16
// [64Mi       ,128Mi)  kv_bf16  -> reused as update_bf16
// [128Mi      ,256Mi)  Y f32    -> reused as A_bf16 -> reused as Yproj f32
// [256Mi      ,272Mi)  g f32
// [272Mi      ,280Mi)  g_bf16
// [280Mi+...          gate_W bf16, proj_W bf16
static const size_t OFF_QB  = 0;
static const size_t OFF_KVB = 67108864ull;
static const size_t OFF_Y   = 134217728ull;
static const size_t OFF_G   = 268435456ull;
static const size_t OFF_GB  = 285212672ull;
static const size_t OFF_GW  = 293601280ull;
static const size_t OFF_PW  = 294125568ull;

extern "C" void kernel_launch(void* const* d_in, const int* in_sizes, int n_in,
                              void* d_out, int out_size, void* d_ws, size_t ws_size,
                              hipStream_t stream) {
    const float* q   = (const float*)d_in[0];
    const float* kv  = (const float*)d_in[1];
    const float* gW  = (const float*)d_in[2];
    const float* gG  = (const float*)d_in[3];
    const float* gB  = (const float*)d_in[4];
    const float* gRM = (const float*)d_in[5];
    const float* gRV = (const float*)d_in[6];
    const float* pW  = (const float*)d_in[7];
    const float* pG  = (const float*)d_in[8];
    const float* pB  = (const float*)d_in[9];
    const float* pRM = (const float*)d_in[10];
    const float* pRV = (const float*)d_in[11];
    float* out = (float*)d_out;

    char* ws = (char*)d_ws;
    unsigned short* qb  = (unsigned short*)(ws + OFF_QB);
    unsigned short* kvb = (unsigned short*)(ws + OFF_KVB);  // later: update bf16
    float*          Y   = (float*)(ws + OFF_Y);
    unsigned short* Ab  = (unsigned short*)(ws + OFF_Y);    // A over Y region
    float*          g   = (float*)(ws + OFF_G);
    unsigned short* gb  = (unsigned short*)(ws + OFF_GB);
    unsigned short* gwB = (unsigned short*)(ws + OFF_GW);
    unsigned short* pwB = (unsigned short*)(ws + OFF_PW);

    const int NBIG = 8 * 8 * 1024 * 512;      // 33,554,432
    const int NW   = 512 * 512;               // 262,144
    const int NELT = 8 * 1024 * 512;          // 4,194,304 (B*N*D)

    f32_to_bf16_kernel<<<NBIG / 256, 256, 0, stream>>>(q,  qb,  NBIG);
    f32_to_bf16_kernel<<<NBIG / 256, 256, 0, stream>>>(kv, kvb, NBIG);
    f32_to_bf16_kernel<<<NW / 256,   256, 0, stream>>>(gW, gwB, NW);
    f32_to_bf16_kernel<<<NW / 256,   256, 0, stream>>>(pW, pwB, NW);

    // GEMM1: Y[65536,512] = kv_bf16 x gate_W^T ; 16384 waves -> 2048 blocks
    gemm_bf16_f32<<<2048, 256, 0, stream>>>(kvb, gwB, Y, 65536, 512, 512);
    gate_lif_kernel<<<NELT / 256, 256, 0, stream>>>(Y, gG, gB, gRM, gRV, g, gb);

    // GEMM2: A = (q x g^T)*scale, bf16, batched over 64; 32768 waves -> 4096 blocks
    gemm_qg_bf16<<<4096, 256, 0, stream>>>(qb, gb, Ab, 0.04419417382415922f);

    // top-4 + sparse update -> bf16 (over kv_bf16 region); 65536 rows, 8 waves/block
    topk_update_kernel<<<8192, 256, 0, stream>>>(Ab, g, kvb);

    // GEMM4: Yp[65536,512] = update_bf16 x proj_W^T (over A region)
    gemm_bf16_f32<<<2048, 256, 0, stream>>>(kvb, pwB, Y, 65536, 512, 512);
    proj_lif_kernel<<<NELT / 256, 256, 0, stream>>>(Y, pG, pB, pRM, pRV, out);
}
